// MoELayerParallel_85040352461278
// MI455X (gfx1250) — matrix operand order repaired and checked by reading, not compiled, not run on hardware
//
#include <hip/hip_runtime.h>

// ---------------------------------------------------------------------------
// MoE top-2 layer for MI455X (gfx1250, wave32).
//  - grouped sparse routing (top-2), per-expert token buckets
//  - fused up-proj (WMMA bf16) -> SiLU -> down-proj (WMMA bf16)
//  - w1/w2 slabs streamed into LDS by the Tensor Data Mover (TDM),
//    double-buffered and overlapped with WMMA via TENSORcnt
//  - fp32 -> bf16 via native hardware conversion (v_cvt_pk_bf16_f32)
// ---------------------------------------------------------------------------

#define E_        8
#define D_        1024
#define H_        4096
#define T_        2048            // BATCH * SEQ
#define MTILE     32              // tokens per expert tile
#define HC        64              // H-chunk (down-proj K slab)
#define KC        128             // D slice staged to LDS for up-proj
#define MAX_TILES (T_ / MTILE)    // 64
#define NSLAB     ((H_ / HC) * (D_ / KC))   // 512 staged slabs per block

typedef __attribute__((ext_vector_type(16))) __bf16 v16bf;
typedef __attribute__((ext_vector_type(8)))  float  v8f;
typedef __attribute__((ext_vector_type(4)))  unsigned int u32x4;
typedef __attribute__((ext_vector_type(8)))  int i32x8;
typedef __attribute__((ext_vector_type(4)))  int i32x4;

#if defined(__has_builtin)
#if __has_builtin(__builtin_amdgcn_tensor_load_to_lds)
#define HAVE_TDM 1
#else
#define HAVE_TDM 0
#endif
#if __has_builtin(__builtin_amdgcn_s_wait_tensorcnt)
#define TENSOR_WAIT(n) __builtin_amdgcn_s_wait_tensorcnt(n)
#else
#define TENSOR_WAIT(n) asm volatile("s_wait_tensorcnt %0" ::"i"(n) : "memory")
#endif
#else
#define HAVE_TDM 0
#define TENSOR_WAIT(n)
#endif

// Native truncation: lowers to hardware packed f32->bf16 conversion on gfx1250.
#define F2BF(f) ((__bf16)(f))

__device__ __forceinline__ unsigned lds_off(const void* p) {
  return (unsigned)(size_t)p;                  // low 32 bits = LDS byte address
}

#if HAVE_TDM
// TDM 2-D tile load: tileH x tileW fp32 elements, source row stride srcStride
// elements, packed row-major into LDS at lds_addr. D# per CDNA5 ISA §8.3/8.4.
__device__ __forceinline__ void tdm_load_2d_f32(unsigned lds_addr,
                                                const float* gptr,
                                                unsigned tileW, unsigned tileH,
                                                unsigned srcStride) {
  unsigned long long ga = (unsigned long long)(size_t)gptr;
  u32x4 g0;
  g0[0] = 1u;                                            // count=1 (user D#)
  g0[1] = lds_addr;                                      // lds_addr [63:32]
  g0[2] = (unsigned)(ga & 0xFFFFFFFFu);                  // global_addr lo
  g0[3] = (unsigned)((ga >> 32) & 0x01FFFFFFu) | (2u << 30); // addr hi | type=2

  unsigned td0 = srcStride;                              // tensor_dim0
  unsigned td1 = tileH;                                  // tensor_dim1
  i32x8 g1;
  g1[0] = (int)(2u << 16);                               // data_size=4B
  g1[1] = (int)((td0 & 0xFFFFu) << 16);                  // dim0[15:0]
  g1[2] = (int)((td0 >> 16) | ((td1 & 0xFFFFu) << 16));  // dim0[31:16]|dim1[15:0]
  g1[3] = (int)((td1 >> 16) | (tileW << 16));            // dim1[31:16]|tile_dim0
  g1[4] = (int)(tileH);                                  // tile_dim1, tile_dim2=0
  g1[5] = (int)(srcStride);                              // dim0_stride lo32
  g1[6] = 0;                                             // stride hi | dim1_stride
  g1[7] = 0;

  i32x4 z4 = {0, 0, 0, 0};                               // 2-D: groups 2/3 NULL-ish
#if __clang_major__ >= 23
  i32x8 z8 = {0, 0, 0, 0, 0, 0, 0, 0};
  __builtin_amdgcn_tensor_load_to_lds(g0, g1, z4, z4, z8, 0);
#else
  __builtin_amdgcn_tensor_load_to_lds(g0, g1, z4, z4, 0);
#endif
}
#endif

// A-fragment (16x32 bf16, M x K) from row-major bf16 LDS.
// Lanes 0-15 hold K 0..15, lanes 16-31 hold K 16..31 (matches B fragments).
__device__ __forceinline__ v16bf load_a16x32(const __bf16* __restrict__ base,
                                             int row0, int ld, int col0, int lane) {
  const bool hi = lane >= 16;
  const __bf16* p = base + (size_t)(row0 + (lane & 15)) * ld + col0;
  v16bf v;
#pragma unroll
  for (int i = 0; i < 16; ++i) {
    int k = (hi ? 16 : 0) + i;
    v[i] = p[k];
  }
  return v;
}

// B-fragment (32x16, K x N) from an N-major [N][K] fp32 LDS tile; hw-convert.
// Lanes 0-15 carry K=0..15, lanes 16-31 carry K=16..31.
__device__ __forceinline__ v16bf load_b32x16_f32(const float* __restrict__ base,
                                                 int row0, int ld, int col0, int lane) {
  const int kb = (lane >= 16) ? 16 : 0;
  const float* p = base + (size_t)(row0 + (lane & 15)) * ld + col0 + kb;
  v16bf v;
#pragma unroll
  for (int i = 0; i < 16; ++i) v[i] = F2BF(p[i]);
  return v;
}

// B-fragment straight from global fp32 (wp rows, each element used once/block).
__device__ __forceinline__ v16bf load_b_global(const float* __restrict__ p, int lane) {
  const int kb = (lane >= 16) ? 16 : 0;
  v16bf v;
#pragma unroll
  for (int i = 0; i < 16; ++i) v[i] = F2BF(p[kb + i]);
  return v;
}

// ---------------------------------------------------------------------------
// Kernel 0: zero output + routing scratch.
// ---------------------------------------------------------------------------
__global__ void moe_init(float* __restrict__ y, int* __restrict__ cnt,
                         float* __restrict__ gw_sum) {
  size_t i = (size_t)blockIdx.x * blockDim.x + threadIdx.x;
  if (i < E_) { cnt[i] = 0; gw_sum[i] = 0.f; }
  size_t stride = (size_t)gridDim.x * blockDim.x;
  for (size_t j = i; j < (size_t)T_ * D_; j += stride) y[j] = 0.f;
}

// ---------------------------------------------------------------------------
// Kernel 1: gating. One wave32 per token.
// ---------------------------------------------------------------------------
__global__ __launch_bounds__(256) void moe_gate(
    const float* __restrict__ x, const float* __restrict__ noise,
    const float* __restrict__ gate_w, const float* __restrict__ noise_w,
    int* __restrict__ cnt, float* __restrict__ gw_sum,
    int* __restrict__ tok_list, float* __restrict__ wgt_list) {
  const int lane = threadIdx.x & 31;
  const int t = blockIdx.x * 8 + (threadIdx.x >> 5);
  if (t >= T_) return;

  float acc[E_];
#pragma unroll
  for (int e = 0; e < E_; ++e) acc[e] = 0.f;
  for (int d = lane; d < D_; d += 32) {
    float xv = x[(size_t)t * D_ + d];
#pragma unroll
    for (int e = 0; e < E_; ++e) acc[e] += xv * gate_w[e * D_ + d];
  }
#pragma unroll
  for (int off = 16; off > 0; off >>= 1) {
#pragma unroll
    for (int e = 0; e < E_; ++e) acc[e] += __shfl_xor(acc[e], off, 32);
  }

  if (lane == 0) {
    float mx = acc[0];
#pragma unroll
    for (int e = 1; e < E_; ++e) mx = fmaxf(mx, acc[e]);
    float ex[E_]; float s = 0.f;
#pragma unroll
    for (int e = 0; e < E_; ++e) { ex[e] = __expf(acc[e] - mx); s += ex[e]; }
    float inv = 1.f / s;
#pragma unroll
    for (int e = 0; e < E_; ++e)
      atomicAdd(&gw_sum[e], ex[e] * inv * (1.f / (float)T_));

    // noisy logits (NOISY_STD = 1.0), top-2 with lowest-index tie-break
    float ln[E_];
#pragma unroll
    for (int e = 0; e < E_; ++e)
      ln[e] = acc[e] + noise[(size_t)t * E_ + e] * 1.0f * noise_w[e];
    int i1 = 0; float v1 = ln[0];
#pragma unroll
    for (int e = 1; e < E_; ++e) if (ln[e] > v1) { v1 = ln[e]; i1 = e; }
    int i2 = -1; float v2 = -3.4e38f;
#pragma unroll
    for (int e = 0; e < E_; ++e) if (e != i1 && ln[e] > v2) { v2 = ln[e]; i2 = e; }

    float e2 = __expf(v2 - v1);
    float w1g = 1.f / (1.f + e2);
    float w2g = e2 / (1.f + e2);

    int p = atomicAdd(&cnt[i1], 1);
    tok_list[i1 * T_ + p] = t; wgt_list[i1 * T_ + p] = w1g;
    p = atomicAdd(&cnt[i2], 1);
    tok_list[i2 * T_ + p] = t; wgt_list[i2 * T_ + p] = w2g;
  }
}

// ---------------------------------------------------------------------------
// Kernel 2: load-balance loss scalar.
// ---------------------------------------------------------------------------
__global__ void moe_loss(const float* __restrict__ gw_sum, float* __restrict__ loss) {
  if (threadIdx.x == 0 && blockIdx.x == 0) {
    float s = 0.f;
#pragma unroll
    for (int e = 0; e < E_; ++e) {
      float d = gw_sum[e] - 1.f / (float)E_;
      s += d * d;
    }
    *loss = (s / (float)E_) * 0.01f;
  }
}

// ---------------------------------------------------------------------------
// Kernel 3: fused expert GEMMs. 256 threads = 8 waves; MTILE=32 tokens/block.
// TDM double-buffers w1/w2 fp32 slabs into LDS while WMMA consumes the
// previous slab; fragments are hw-converted to bf16 at load time.
// ---------------------------------------------------------------------------
__global__ __launch_bounds__(256) void moe_expert(
    const float* __restrict__ x,
    const float* __restrict__ w1, const float* __restrict__ b1,
    const float* __restrict__ w2, const float* __restrict__ b2,
    const float* __restrict__ wp, const float* __restrict__ bp,
    const int* __restrict__ cnt, const int* __restrict__ tok_list,
    const float* __restrict__ wgt_list, float* __restrict__ y) {
  __shared__ __bf16 xs[MTILE * D_];             // 64 KB token tile (bf16)
  __shared__ float wstage[2][2][HC * KC];       // 128 KB double-buffered w1/w2 slabs
  __shared__ __bf16 acts[MTILE * HC];           //  4 KB activation tile (bf16)
  __shared__ int   s_tok[MTILE];
  __shared__ float s_wgt[MTILE];

  const int e    = blockIdx.x / MAX_TILES;
  const int tile = blockIdx.x % MAX_TILES;
  const int nrows = cnt[e];
  if (tile * MTILE >= nrows) return;

  const int tid  = threadIdx.x;
  const int lane = tid & 31;
  const int wave = tid >> 5;

  if (tid < MTILE) {
    int r = tile * MTILE + tid;
    if (r < nrows) {
      s_tok[tid] = tok_list[e * T_ + r];
      s_wgt[tid] = wgt_list[e * T_ + r];
    } else {
      s_tok[tid] = 0;      // padded row: weight 0 kills its contribution
      s_wgt[tid] = 0.f;
    }
  }
  __syncthreads();

  const float* w1e = w1 + (size_t)e * H_ * D_;
  const float* w2e = w2 + (size_t)e * H_ * D_;

#if HAVE_TDM
  // Prologue: wave 0 kicks the TDM on slab 0 while everyone stages xs.
  if (wave == 0) {
    tdm_load_2d_f32(lds_off(&wstage[0][0][0]), w1e, KC, HC, D_);
    tdm_load_2d_f32(lds_off(&wstage[0][1][0]), w2e, KC, HC, D_);
  }
#endif

  // gather + hw-convert token rows into LDS (bf16)
#pragma unroll 1
  for (int i = tid; i < MTILE * D_; i += 256) {
    int m = i >> 10;
    int d = i & (D_ - 1);
    xs[i] = F2BF(x[(size_t)s_tok[m] * D_ + d]);
  }

  const int nbase = wave * 128;       // wave's slice of output columns
  v8f outc[16];                       // 2 M-tiles x 8 N-tiles
  const v8f vzero = {0.f, 0.f, 0.f, 0.f, 0.f, 0.f, 0.f, 0.f};
#pragma unroll
  for (int i = 0; i < 16; ++i) outc[i] = vzero;

  const int mt_up = wave >> 2;        // 16-token half for up-proj
  const int nt_up = wave & 3;         // 16-wide H column for up-proj

#pragma unroll 1
  for (int hc = 0; hc < H_ / HC; ++hc) {
    v8f hacc = vzero, gacc = vzero;

    // ---- up-projections h = x@w1^T, g = x@w2^T over K = D in KC slabs ----
#pragma unroll 1
    for (int kc = 0; kc < D_ / KC; ++kc) {
      const int s = hc * (D_ / KC) + kc;
      const int buf = s & 1;
      __syncthreads();                // all readers done with buf[s+1 parity]
#if HAVE_TDM
      if (wave == 0) {
        if (s + 1 < NSLAB) {          // stream next slab while computing this one
          int hn = (s + 1) / (D_ / KC), kn = (s + 1) % (D_ / KC);
          tdm_load_2d_f32(lds_off(&wstage[buf ^ 1][0][0]),
                          w1e + (size_t)hn * HC * D_ + (size_t)kn * KC, KC, HC, D_);
          tdm_load_2d_f32(lds_off(&wstage[buf ^ 1][1][0]),
                          w2e + (size_t)hn * HC * D_ + (size_t)kn * KC, KC, HC, D_);
          TENSOR_WAIT(2);             // pair for slab s complete (in-order)
        } else {
          TENSOR_WAIT(0);
        }
      }
#else
      // fallback: cooperative fp32 copy of slab s
      {
        const float* w1p = w1e + (size_t)hc * HC * D_ + (size_t)kc * KC;
        const float* w2p = w2e + (size_t)hc * HC * D_ + (size_t)kc * KC;
#pragma unroll 1
        for (int i = tid; i < HC * KC; i += 256) {
          int r = i >> 7, c = i & (KC - 1);
          wstage[buf][0][i] = w1p[(size_t)r * D_ + c];
          wstage[buf][1][i] = w2p[(size_t)r * D_ + c];
        }
      }
#endif
      __syncthreads();                // buf[s] published to all waves
#pragma unroll
      for (int kk = 0; kk < KC; kk += 32) {
        v16bf a  = load_a16x32(xs, mt_up * 16, D_, kc * KC + kk, lane);
        v16bf bh = load_b32x16_f32(wstage[buf][0], nt_up * 16, KC, kk, lane);
        v16bf bg = load_b32x16_f32(wstage[buf][1], nt_up * 16, KC, kk, lane);
        hacc = __builtin_amdgcn_wmma_f32_16x16x32_bf16(false, a, false, bh,
                                                       (short)0, hacc, false, false);
        gacc = __builtin_amdgcn_wmma_f32_16x16x32_bf16(false, a, false, bg,
                                                       (short)0, gacc, false, false);
      }
    }

    // ---- bias + h * SiLU(g) -> act tile (bf16 LDS) ----
    {
      int ncol = hc * HC + nt_up * 16 + (lane & 15);
      float bhv = b1[(size_t)e * H_ + ncol];
      float bgv = b2[(size_t)e * H_ + ncol];
      int mrow0 = mt_up * 16 + ((lane >= 16) ? 8 : 0);
      int nloc  = nt_up * 16 + (lane & 15);
#pragma unroll
      for (int r = 0; r < 8; ++r) {
        float hv = hacc[r] + bhv;
        float gv = gacc[r] + bgv;
        float sg = gv / (1.f + __expf(-gv));         // SiLU
        acts[(mrow0 + r) * HC + nloc] = F2BF(hv * sg);
      }
      __syncthreads();
    }

    // ---- down-projection: outc += act @ wp^T (wp streamed from global) ----
    const float* wpe = wp + (size_t)e * D_ * H_;
#pragma unroll
    for (int kk = 0; kk < HC; kk += 32) {
      v16bf a0 = load_a16x32(acts, 0,  HC, kk, lane);
      v16bf a1 = load_a16x32(acts, 16, HC, kk, lane);
#pragma unroll 1
      for (int nt2 = 0; nt2 < 8; ++nt2) {
        const float* wpp = wpe +
            (size_t)(nbase + nt2 * 16 + (lane & 15)) * H_ + (size_t)hc * HC + kk;
        v16bf bfrag = load_b_global(wpp, lane);
        outc[nt2] = __builtin_amdgcn_wmma_f32_16x16x32_bf16(
            false, a0, false, bfrag, (short)0, outc[nt2], false, false);
        outc[8 + nt2] = __builtin_amdgcn_wmma_f32_16x16x32_bf16(
            false, a1, false, bfrag, (short)0, outc[8 + nt2], false, false);
      }
    }
    // next chunk's first __syncthreads() fences acts + slab reuse
  }

  // ---- epilogue: gate-weighted scatter with bp bias ----
  const int nlow = lane & 15;
  const int mofs = (lane >= 16) ? 8 : 0;
#pragma unroll 1
  for (int mt2 = 0; mt2 < 2; ++mt2) {
#pragma unroll 1
    for (int nt2 = 0; nt2 < 8; ++nt2) {
      int dcol = nbase + nt2 * 16 + nlow;
      float bpv = bp[(size_t)e * D_ + dcol];
#pragma unroll
      for (int r = 0; r < 8; ++r) {
        int m = mt2 * 16 + mofs + r;
        float v = outc[mt2 * 8 + nt2][r] + bpv;
        atomicAdd(&y[(size_t)s_tok[m] * D_ + dcol], s_wgt[m] * v);
      }
    }
  }
}

// ---------------------------------------------------------------------------
// Host entry
// ---------------------------------------------------------------------------
extern "C" void kernel_launch(void* const* d_in, const int* in_sizes, int n_in,
                              void* d_out, int out_size, void* d_ws, size_t ws_size,
                              hipStream_t stream) {
  const float* x       = (const float*)d_in[0];
  const float* noise   = (const float*)d_in[1];
  const float* gate_w  = (const float*)d_in[2];
  const float* noise_w = (const float*)d_in[3];
  const float* w1      = (const float*)d_in[4];
  const float* b1      = (const float*)d_in[5];
  const float* w2      = (const float*)d_in[6];
  const float* b2      = (const float*)d_in[7];
  const float* wp      = (const float*)d_in[8];
  const float* bp      = (const float*)d_in[9];

  float* y    = (float*)d_out;
  float* loss = y + (size_t)T_ * D_;        // scalar is last output element

  // workspace layout (131,200 bytes):
  //   [0,32)       cnt[8]
  //   [64,96)      gw_sum[8]
  //   [128,+64K)   tok_list[E*T]
  //   [+64K,+128K) wgt_list[E*T]
  char* ws = (char*)d_ws;
  int*   cnt      = (int*)ws;
  float* gw_sum   = (float*)(ws + 64);
  int*   tok_list = (int*)(ws + 128);
  float* wgt_list = (float*)(ws + 128 + (size_t)E_ * T_ * sizeof(int));

  moe_init  <<<1024, 256, 0, stream>>>(y, cnt, gw_sum);
  moe_gate  <<<T_ / 8, 256, 0, stream>>>(x, noise, gate_w, noise_w,
                                         cnt, gw_sum, tok_list, wgt_list);
  moe_loss  <<<1, 32, 0, stream>>>(gw_sum, loss);
  moe_expert<<<E_ * MAX_TILES, 256, 0, stream>>>(x, w1, b1, w2, b2, wp, bp,
                                                 cnt, tok_list, wgt_list, y);
}
